// QKVAttention_7456063226403
// MI455X (gfx1250) — matrix operand order repaired and checked by reading, not compiled, not run on hardware
//
#include <hip/hip_runtime.h>
#include <hip/hip_bf16.h>
#include <math.h>

typedef __attribute__((ext_vector_type(16))) __bf16 v16bf;
typedef __attribute__((ext_vector_type(8)))  float  v8f;

#define NB   64
#define CDIM 128
#define TDIM 2048
#define BM   128          // t rows per workgroup
#define BN   32           // s columns per block
#define CP   136          // padded bf16 stride for [row][c] tiles (272 B, 16B-mult)
#define BNP  40           // padded bf16 stride for [row][s] tiles (80 B, 16B-mult)

// LDS offsets (bytes) within dynamic smem (total 64000 B), all 16B-aligned
#define OFF_Q 0           // BM*CP*2      = 34816
#define OFF_K 34816       // BN*CP*2      =  8704
#define OFF_V 43520       // CDIM*BNP*2   = 10240
#define OFF_P 53760       // 8*16*BNP*2   = 10240

union ABu { uint4 q2[2]; v16bf v; };

static __device__ __forceinline__ unsigned pack2bf(float a, float b) {
  union { __bf16 h[2]; unsigned u; } p;
  p.h[0] = (__bf16)a; p.h[1] = (__bf16)b;
  return p.u;
}

__global__ __launch_bounds__(256) void qkv_attn_wmma(const float* __restrict__ qkv,
                                                     float* __restrict__ out) {
  extern __shared__ char smem[];
  __bf16* sQ = (__bf16*)(smem + OFF_Q);
  __bf16* sK = (__bf16*)(smem + OFF_K);
  __bf16* sV = (__bf16*)(smem + OFF_V);
  __bf16* sP = (__bf16*)(smem + OFF_P);

  const int tid  = threadIdx.x;
  const int wave = tid >> 5;
  const int lane = tid & 31;
  const int half = lane >> 4;          // which 16-lane half
  const int ln   = lane & 15;

  const int b  = blockIdx.y;
  const int t0 = blockIdx.x * BM;

  const float* q = qkv + (size_t)b * 3 * CDIM * TDIM;
  const float* k = q + (size_t)CDIM * TDIM;
  const float* v = q + (size_t)2 * CDIM * TDIM;

  // 128^(-1/4) * sqrt(log2(e)) applied to q and k -> softmax uses bare exp2
  const float scale = 0.35709582847f;

  // ---- stage Q tile: LDS [t_local][c] (transpose from [c][t]), float4 reads ----
  for (int idx = tid; idx < BM * CDIM / 4; idx += 256) {
    int c  = idx >> 5;                  // / (BM/4)
    int i4 = idx & 31;
    float4 val = *(const float4*)(q + (size_t)c * TDIM + t0 + i4 * 4);
    sQ[(i4 * 4 + 0) * CP + c] = (__bf16)(val.x * scale);
    sQ[(i4 * 4 + 1) * CP + c] = (__bf16)(val.y * scale);
    sQ[(i4 * 4 + 2) * CP + c] = (__bf16)(val.z * scale);
    sQ[(i4 * 4 + 3) * CP + c] = (__bf16)(val.w * scale);
  }

  v8f o[8];                            // O'[c,t]: M=c tile rows, N=t columns
  #pragma unroll
  for (int i = 0; i < 8; ++i) { v8f z = {}; o[i] = z; }
  float mrun = -INFINITY, lrun = 0.f;  // per-lane stats for column t = ln

  __bf16* pw = sP + wave * 16 * BNP;                        // per-wave P region
  const __bf16* qbase  = sQ + (wave * 16 + ln) * CP + 8 * half;    // Q as B operand
  const __bf16* kbase0 = sK + (0  + ln) * CP + 8 * half;           // K as A, s-tile 0
  const __bf16* kbase1 = sK + (16 + ln) * CP + 8 * half;           // K as A, s-tile 1
  const uint4*  pbase  = (const uint4*)(pw + ln * BNP + 8 * half); // P as B operand

  // per-thread staging geometry: j4 constant, c = c0 + 32*i
  const int j4 = tid & 7;
  const int c0 = tid >> 3;
  const float* kptr = k + (size_t)c0 * TDIM + j4 * 4;
  const float* vptr = v + (size_t)c0 * TDIM + j4 * 4;

  float4 kf[4], vf[4];
  #pragma unroll
  for (int i = 0; i < 4; ++i) {        // prefetch block 0
    kf[i] = *(const float4*)(kptr + (size_t)(32 * i) * TDIM);
    vf[i] = *(const float4*)(vptr + (size_t)(32 * i) * TDIM);
  }

  for (int sb = 0; sb < TDIM / BN; ++sb) {
    __syncthreads();   // previous block's LDS reads done before overwrite
    // store staged K/V fragments: sK[s_local][c] (transposed), sV[c][s_local]
    #pragma unroll
    for (int i = 0; i < 4; ++i) {
      int c = c0 + 32 * i;
      sK[(j4 * 4 + 0) * CP + c] = (__bf16)(kf[i].x * scale);
      sK[(j4 * 4 + 1) * CP + c] = (__bf16)(kf[i].y * scale);
      sK[(j4 * 4 + 2) * CP + c] = (__bf16)(kf[i].z * scale);
      sK[(j4 * 4 + 3) * CP + c] = (__bf16)(kf[i].w * scale);
      uint2 pkd;
      pkd.x = pack2bf(vf[i].x, vf[i].y);
      pkd.y = pack2bf(vf[i].z, vf[i].w);
      *(uint2*)(sV + c * BNP + j4 * 4) = pkd;
    }
    // prefetch next block's K/V into registers (latency hidden by this block's WMMAs)
    if (sb + 1 < TDIM / BN) {
      const int s1 = (sb + 1) * BN;
      #pragma unroll
      for (int i = 0; i < 4; ++i) {
        kf[i] = *(const float4*)(kptr + (size_t)(32 * i) * TDIM + s1);
        vf[i] = *(const float4*)(vptr + (size_t)(32 * i) * TDIM + s1);
      }
    }
    __syncthreads();

    // ---- S^T = K Q^T : two 16(s)x16(t) tiles, K-dim = C = 4 x 32 ----
    v8f sacc[2];
    { v8f z = {}; sacc[0] = z; sacc[1] = z; }
    #pragma unroll
    for (int kc = 0; kc < 4; ++kc) {
      ABu B;
      const uint4* qrow4 = (const uint4*)(qbase + kc * 32);
      B.q2[0] = qrow4[0];        // K(c) = 8*half + 0..7
      B.q2[1] = qrow4[2];        // K(c) = 16 + 8*half + 0..7
      ABu A0, A1;
      const uint4* k0 = (const uint4*)(kbase0 + kc * 32);
      const uint4* k1 = (const uint4*)(kbase1 + kc * 32);
      A0.q2[0] = k0[0]; A0.q2[1] = k0[2];   // K(c) = 8*half / 16+8*half
      A1.q2[0] = k1[0]; A1.q2[1] = k1[2];
      sacc[0] = __builtin_amdgcn_wmma_f32_16x16x32_bf16(
          false, A0.v, false, B.v, (short)0, sacc[0], false, false);
      sacc[1] = __builtin_amdgcn_wmma_f32_16x16x32_bf16(
          false, A1.v, false, B.v, (short)0, sacc[1], false, false);
    }

    // ---- online softmax: reduction axis s == M axis -> lane-local ----
    float xm = fmaxf(sacc[0][0], sacc[1][0]);
    #pragma unroll
    for (int r = 1; r < 8; ++r) xm = fmaxf(xm, fmaxf(sacc[0][r], sacc[1][r]));
    xm = fmaxf(xm, __shfl_xor(xm, 16, 32));      // combine the two halves
    const bool maxup = __ballot(xm > mrun) != 0ull;  // wave-uniform
    float mnew = fmaxf(mrun, xm);
    float psum = 0.f;
    #pragma unroll
    for (int nsub = 0; nsub < 2; ++nsub)
      #pragma unroll
      for (int r = 0; r < 8; ++r) {
        float p = exp2f(sacc[nsub][r] - mnew);   // (s = nsub*16+r+8*half, t = ln)
        pw[ln * BNP + nsub * 16 + r + 8 * half] = (__bf16)p;
        psum += p;
      }
    psum += __shfl_xor(psum, 16, 32);
    if (maxup) {                                 // rescale only when max moved
      float alpha = exp2f(mrun - mnew);
      lrun = lrun * alpha + psum;
      mrun = mnew;
      #pragma unroll
      for (int ct = 0; ct < 8; ++ct)
        #pragma unroll
        for (int r = 0; r < 8; ++r) o[ct][r] *= alpha;
    } else {
      lrun += psum;
    }

    // ---- O' += V P : 8 c-tiles (A = V), B = P fixed ----
    ABu Bp;
    Bp.q2[0] = pbase[0];          // K(s) = 8*half + 0..7
    Bp.q2[1] = pbase[2];          // K(s) = 16 + 8*half + 0..7
    #pragma unroll
    for (int ct = 0; ct < 8; ++ct) {
      ABu Av;
      const uint4* vrow4 = (const uint4*)(sV + (ct * 16 + ln) * BNP + 8 * half);
      Av.q2[0] = vrow4[0];        // K(s) = 8*half + 0..7
      Av.q2[1] = vrow4[2];        // K(s) = 16 + 8*half + 0..7
      o[ct] = __builtin_amdgcn_wmma_f32_16x16x32_bf16(
          false, Av.v, false, Bp.v, (short)0, o[ct], false, false);
    }
  }

  // ---- normalize and store: D-tile is already [c][t]; t = lane -> coalesced ----
  float inv = 1.f / lrun;
  float* outp = out + (size_t)b * CDIM * TDIM + t0 + wave * 16 + ln;
  #pragma unroll
  for (int ct = 0; ct < 8; ++ct)
    #pragma unroll
    for (int r = 0; r < 8; ++r)
      outp[(size_t)(ct * 16 + r + 8 * half) * TDIM] = o[ct][r] * inv;
}

extern "C" void kernel_launch(void* const* d_in, const int* in_sizes, int n_in,
                              void* d_out, int out_size, void* d_ws, size_t ws_size,
                              hipStream_t stream) {
  (void)in_sizes; (void)n_in; (void)d_ws; (void)ws_size; (void)out_size;
  const float* qkv = (const float*)d_in[0];
  float* out = (float*)d_out;
  dim3 grid(TDIM / BM, NB);
  size_t smem = 64000;   // Q + K + V + P regions
  qkv_attn_wmma<<<grid, 256, smem, stream>>>(qkv, out);
}
